// GaelicMorphModel_2851858284917
// MI455X (gfx1250) — compile-verified, matrix operand order checked by reading, not run on hardware
//
#include <hip/hip_runtime.h>
#include <hip/hip_bf16.h>
#include <math.h>

// ---------------- model dims ----------------
#define VW   50000
#define VC   200
#define TAGS 600
#define WE   128
#define CE   48
#define CC   64
#define HDIM 256
#define BB   64
#define TT   96
#define CL   16
#define LMAX 24
#define HD   128          // per-direction LSTM hidden
#define NROWS (BB*TT)     // 6144
#define DIN  (WE+CC)      // 192
#define BOSTOK 3

typedef _Float16 half_t;
typedef __attribute__((ext_vector_type(16))) _Float16 v16h;
typedef __attribute__((ext_vector_type(8)))  _Float16 v8h;
typedef __attribute__((ext_vector_type(8)))  float    v8f;

// ---------------- WMMA fragment loader ----------------
// A (16x32 f16) layout, wave32: lane l -> row M=l&15, half h=l>>4.
// element j<8  : K = k0 + 8h + j          (contiguous 8 halves)
// element j>=8 : K = k0 + 16 + 8h + (j-8) (contiguous 8 halves)
// B (32x16) uses the identical per-lane layout with N in place of M,
// i.e. load row (n0+lane&15) of the [N x K] weight matrix.
__device__ __forceinline__ v16h load_frag(const half_t* base, int row0, int ld,
                                          int k0, int lane) {
  int m  = lane & 15;
  int hh = lane >> 4;
  const half_t* p = base + (size_t)(row0 + m) * ld + k0 + 8 * hh;
  v8h lo = *(const v8h*)(p);
  v8h hi = *(const v8h*)(p + 16);
  v16h r;
#pragma unroll
  for (int i = 0; i < 8; ++i) { r[i] = lo[i]; r[i + 8] = hi[i]; }
  return r;
}

__device__ __forceinline__ float sigf(float x) { return 1.0f / (1.0f + expf(-x)); }

// ---------------- generic WMMA GEMM: C = A(f16) @ B(f16)^T + bias ----------------
// Register-blocked: each wave computes a 16x64 output block (A fragment reused
// across 4 B fragments). A:[M x K] f16, B:[Npad x K] f16 (torch weight layout),
// C:[M x ldC] f32, store cols < Nstore. M%16==0, Npad%64==0, K%32==0.
__global__ __launch_bounds__(32) void wmma_gemm_k(const half_t* __restrict__ A,
                                                  const half_t* __restrict__ B,
                                                  const float* __restrict__ bias,
                                                  float* __restrict__ C,
                                                  int M, int Npad, int K,
                                                  int ldC, int Nstore) {
  int groupsN = Npad >> 6;                 // 4 tiles of 16 per wave
  int rt = blockIdx.x / groupsN;
  int cg = blockIdx.x % groupsN;
  int lane = threadIdx.x;
  v8f acc[4] = {};
  for (int k0 = 0; k0 < K; k0 += 32) {
    if (k0 + 32 < K)
      __builtin_prefetch(B + (size_t)(cg * 64) * K + k0 + 32, 0, 1);
    v16h a = load_frag(A, rt * 16, K, k0, lane);
#pragma unroll
    for (int j = 0; j < 4; ++j) {
      v16h b = load_frag(B, (cg * 4 + j) * 16, K, k0, lane);
      acc[j] = __builtin_amdgcn_wmma_f32_16x16x32_f16(false, a, false, b, (short)0,
                                                      acc[j], false, false);
    }
  }
  int n  = lane & 15;
  int hh = lane >> 4;
#pragma unroll
  for (int j = 0; j < 4; ++j) {
    int col = (cg * 4 + j) * 16 + n;
    if (col < Nstore) {
      float bv = bias ? bias[col] : 0.0f;
#pragma unroll
      for (int r = 0; r < 8; ++r) {
        int row = rt * 16 + hh * 8 + r;
        C[(size_t)row * ldC + col] = acc[j][r] + bv;
      }
    }
  }
}

// ---------------- f32 -> f16 convert with zero padding ----------------
__global__ void cvt_pad16_k(const float* __restrict__ src, half_t* __restrict__ dst,
                            int rows, int cols, int srows, int scols) {
  int i = blockIdx.x * blockDim.x + threadIdx.x;
  if (i >= rows * cols) return;
  int r = i / cols, c = i % cols;
  float v = (r < srows && c < scols) ? src[r * scols + c] : 0.0f;
  dst[i] = (half_t)v;
}

__global__ void pad_bias_k(const float* __restrict__ src, float* __restrict__ dst,
                           int n, int sn, float padv) {
  int i = blockIdx.x * blockDim.x + threadIdx.x;
  if (i >= n) return;
  dst[i] = (i < sn) ? src[i] : padv;
}

// ---------------- embeddings + char conv (K=3 SAME) + relu + maxpool ----------------
// one block per word position n (0..6143), 64 threads (one per conv out channel)
__global__ __launch_bounds__(64) void embed_conv_k(const int* __restrict__ word_ids,
                                                   const int* __restrict__ char_ids,
                                                   const float* __restrict__ word_emb,
                                                   const float* __restrict__ char_emb,
                                                   const float* __restrict__ cw,
                                                   const float* __restrict__ cb,
                                                   half_t* __restrict__ enc_in16) {
  __shared__ __align__(16) float sce[CL][CE];
  __shared__ __align__(16) float sw[CC * CE * 3];
  __shared__ int scid[CL];
  int n = blockIdx.x, tid = threadIdx.x;
  for (int i = tid; i < CC * CE * 3; i += 64) sw[i] = cw[i];
  if (tid < CL) scid[tid] = char_ids[(size_t)n * CL + tid];
  __syncthreads();
  for (int i = tid; i < CL * CE; i += 64) {
    int l = i / CE, c = i % CE;
    sce[l][c] = char_emb[(size_t)scid[l] * CE + c];
  }
  int wid = word_ids[n];
  enc_in16[(size_t)n * DIN + tid]      = (half_t)word_emb[(size_t)wid * WE + tid];
  enc_in16[(size_t)n * DIN + tid + 64] = (half_t)word_emb[(size_t)wid * WE + tid + 64];
  __syncthreads();
  int cc = tid;
  float best = -1e30f;
  for (int l = 0; l < CL; ++l) {
    float acc = cb[cc];
#pragma unroll
    for (int dl = 0; dl < 3; ++dl) {
      int lp = l + dl - 1;
      if (lp < 0 || lp >= CL) continue;
      const float* wrow = &sw[cc * (CE * 3) + dl];
      for (int c = 0; c < CE; ++c) acc += sce[lp][c] * wrow[c * 3];
    }
    best = fmaxf(best, acc);
  }
  enc_in16[(size_t)n * DIN + WE + cc] = (half_t)fmaxf(best, 0.0f);  // relu(max)
}

// ---------------- LSTM recurrence (both directions) ----------------
// grid: 8 blocks = 2 dirs x 4 batch-groups of 16 rows; 128 threads (4 waves).
// Gx (x@Wih^T + b) precomputed; per step g = h@Whh^T + Gx_t via WMMA.
// A fragments (h) preloaded into registers once per step, reused over 8 tiles.
__global__ __launch_bounds__(128) void lstm_rec_k(const half_t* __restrict__ whh_fw,
                                                  const half_t* __restrict__ whh_bw,
                                                  const float* __restrict__ gx_fw,
                                                  const float* __restrict__ gx_bw,
                                                  half_t* __restrict__ enc16) {
  __shared__ __align__(16) half_t sh[16 * HD];       // h, f16
  __shared__ __align__(16) float  sc[16 * HD];       // c, f32
  __shared__ __align__(16) half_t sg[16 * 4 * HD];   // gates, f16
  int dir = blockIdx.x >> 2;
  int b0  = (blockIdx.x & 3) * 16;
  const half_t* whh = dir ? whh_bw : whh_fw;
  const float*  gx  = dir ? gx_bw : gx_fw;
  int tid = threadIdx.x, wave = tid >> 5, lane = tid & 31;
  int nn = lane & 15, hh = lane >> 4;
  for (int i = tid; i < 16 * HD; i += 128) { sh[i] = (half_t)0.0f; sc[i] = 0.0f; }
  __syncthreads();
  for (int s = 0; s < TT; ++s) {
    int t = dir ? (TT - 1 - s) : s;
    // preload A fragments of h (K=128 -> 4 fragments)
    v16h afr[4];
#pragma unroll
    for (int kk = 0; kk < 4; ++kk) afr[kk] = load_frag(sh, 0, HD, kk * 32, lane);
    // g = h @ Whh^T + Gx[:,t,:]
    for (int ci = 0; ci < 8; ++ci) {
      int ct = wave * 8 + ci;
      v8f acc = {};
#pragma unroll
      for (int kk = 0; kk < 4; ++kk) {
        v16h b = load_frag(whh, ct * 16, HD, kk * 32, lane);
        acc = __builtin_amdgcn_wmma_f32_16x16x32_f16(false, afr[kk], false, b,
                                                     (short)0, acc, false, false);
      }
      int col = ct * 16 + nn;
#pragma unroll
      for (int r = 0; r < 8; ++r) {
        int m = hh * 8 + r;
        size_t gn = (size_t)(b0 + m) * TT + t;
        sg[m * (4 * HD) + col] = (half_t)(acc[r] + gx[gn * (4 * HD) + col]);
      }
    }
    __syncthreads();
    // gates: torch order i,f,g,o
    for (int e = tid; e < 16 * HD; e += 128) {
      int m = e >> 7, j = e & (HD - 1);
      const half_t* g = &sg[m * (4 * HD)];
      float gi = (float)g[j], gf = (float)g[HD + j];
      float gg = (float)g[2 * HD + j], go = (float)g[3 * HD + j];
      float c = sigf(gf) * sc[e] + sigf(gi) * tanhf(gg);
      sc[e] = c;
      float h = sigf(go) * tanhf(c);
      sh[e] = (half_t)h;
      size_t gn = (size_t)(b0 + m) * TT + t;
      enc16[gn * HDIM + dir * HD + j] = (half_t)h;
    }
    __syncthreads();
  }
}

// ---------------- fused greedy GRU lemma decoder ----------------
// one block per 16 rows (384 blocks), 128 threads (4 waves), 24 steps.
// A fragments of h preloaded per phase (8 x v16h) and reused across all
// column tiles the wave owns.
__global__ __launch_bounds__(128) void decoder_k(const half_t* __restrict__ enc16,
                                                 const half_t* __restrict__ init16,
                                                 const float* __restrict__ initb,
                                                 const half_t* __restrict__ ghh16,
                                                 const float* __restrict__ bhh,
                                                 const float* __restrict__ gtab,
                                                 const half_t* __restrict__ proj16,
                                                 const float* __restrict__ projb,
                                                 float* __restrict__ out_lemma) {
  __shared__ __align__(16) half_t sh[16 * HDIM];   // h, f16 (8KB)
  __shared__ __align__(16) float  sg[16 * 768];    // gh / logits scratch (48KB)
  __shared__ int sprev[16];
  int row0 = blockIdx.x * 16;
  int tid = threadIdx.x, wave = tid >> 5, lane = tid & 31;
  int nn = lane & 15, hh = lane >> 4;
  v16h afr[8];

  // h0 = tanh(enc @ initW^T + initb); A fragments from enc16 (global)
#pragma unroll
  for (int kk = 0; kk < 8; ++kk) afr[kk] = load_frag(enc16, row0, HDIM, kk * 32, lane);
  for (int ct = wave; ct < 16; ct += 4) {
    v8f acc = {};
#pragma unroll
    for (int kk = 0; kk < 8; ++kk) {
      v16h b = load_frag(init16, ct * 16, HDIM, kk * 32, lane);
      acc = __builtin_amdgcn_wmma_f32_16x16x32_f16(false, afr[kk], false, b,
                                                   (short)0, acc, false, false);
    }
    int col = ct * 16 + nn;
    float bv = initb[col];
#pragma unroll
    for (int r = 0; r < 8; ++r)
      sh[(hh * 8 + r) * HDIM + col] = (half_t)tanhf(acc[r] + bv);
  }
  if (tid < 16) sprev[tid] = BOSTOK;
  __syncthreads();

  for (int step = 0; step < LMAX; ++step) {
    // gh = h @ Whh^T + bhh  (768 cols); preload h fragments once
#pragma unroll
    for (int kk = 0; kk < 8; ++kk) afr[kk] = load_frag(sh, 0, HDIM, kk * 32, lane);
    for (int ct = wave; ct < 48; ct += 4) {
      if (ct + 4 < 48) __builtin_prefetch(ghh16 + (size_t)((ct + 4) * 16) * HDIM, 0, 1);
      v8f acc = {};
#pragma unroll
      for (int kk = 0; kk < 8; ++kk) {
        v16h b = load_frag(ghh16, ct * 16, HDIM, kk * 32, lane);
        acc = __builtin_amdgcn_wmma_f32_16x16x32_f16(false, afr[kk], false, b,
                                                     (short)0, acc, false, false);
      }
      int col = ct * 16 + nn;
      float bv = bhh[col];
#pragma unroll
      for (int r = 0; r < 8; ++r) sg[(hh * 8 + r) * 768 + col] = acc[r] + bv;
    }
    __syncthreads();
    // GRU gate fusion; gi comes from the precomputed token table
    for (int e = tid; e < 16 * HDIM; e += 128) {
      int m = e >> 8, j = e & (HDIM - 1);
      const float* gt = gtab + (size_t)sprev[m] * 768;
      const float* g  = &sg[m * 768];
      float r_ = sigf(gt[j] + g[j]);
      float z_ = sigf(gt[HDIM + j] + g[HDIM + j]);
      float n_ = tanhf(gt[2 * HDIM + j] + r_ * g[2 * HDIM + j]);
      float ho = (float)sh[e];
      sh[e] = (half_t)((1.0f - z_) * n_ + z_ * ho);
    }
    __syncthreads();
    // logits = h @ proj^T + projb (208 padded cols; pad bias = -1e30)
#pragma unroll
    for (int kk = 0; kk < 8; ++kk) afr[kk] = load_frag(sh, 0, HDIM, kk * 32, lane);
    for (int ct = wave; ct < 13; ct += 4) {
      v8f acc = {};
#pragma unroll
      for (int kk = 0; kk < 8; ++kk) {
        v16h b = load_frag(proj16, ct * 16, HDIM, kk * 32, lane);
        acc = __builtin_amdgcn_wmma_f32_16x16x32_f16(false, afr[kk], false, b,
                                                     (short)0, acc, false, false);
      }
      int col = ct * 16 + nn;
      float bv = projb[col];
#pragma unroll
      for (int r = 0; r < 8; ++r) {
        int m = hh * 8 + r;
        float v = acc[r] + bv;
        sg[m * 208 + col] = v;
        if (col < VC)
          out_lemma[((size_t)(row0 + m) * LMAX + step) * VC + col] = v;
      }
    }
    __syncthreads();
    // per-row greedy argmax (first-max tie-break like jnp.argmax)
    if (tid < 16) {
      const float* lp = &sg[tid * 208];
      float best = lp[0]; int bi = 0;
      for (int v = 1; v < VC; ++v) { float x = lp[v]; if (x > best) { best = x; bi = v; } }
      sprev[tid] = bi;
    }
    __syncthreads();
  }
}

// ---------------- host ----------------
extern "C" void kernel_launch(void* const* d_in, const int* in_sizes, int n_in,
                              void* d_out, int out_size, void* d_ws, size_t ws_size,
                              hipStream_t stream) {
  (void)in_sizes; (void)n_in; (void)out_size; (void)ws_size;
  const int*   word_ids = (const int*)d_in[0];
  const int*   char_ids = (const int*)d_in[1];
  const float* word_emb = (const float*)d_in[2];
  const float* char_emb = (const float*)d_in[3];
  const float* conv_w   = (const float*)d_in[4];
  const float* conv_b   = (const float*)d_in[5];
  const float* fw_wih   = (const float*)d_in[6];
  const float* fw_whh   = (const float*)d_in[7];
  const float* fw_b     = (const float*)d_in[8];
  const float* bw_wih   = (const float*)d_in[9];
  const float* bw_whh   = (const float*)d_in[10];
  const float* bw_b     = (const float*)d_in[11];
  const float* tag_w    = (const float*)d_in[12];
  const float* tag_b    = (const float*)d_in[13];
  const float* lemma_emb  = (const float*)d_in[14];
  const float* init_w   = (const float*)d_in[15];
  const float* init_b   = (const float*)d_in[16];
  const float* gru_wih  = (const float*)d_in[17];
  const float* gru_whh  = (const float*)d_in[18];
  const float* gru_bih  = (const float*)d_in[19];
  const float* gru_bhh  = (const float*)d_in[20];
  const float* proj_w   = (const float*)d_in[21];
  const float* proj_b   = (const float*)d_in[22];

  char* ws = (char*)d_ws;
  size_t off = 0;
  auto alloc = [&](size_t bytes) { size_t o = off; off += (bytes + 255) & ~(size_t)255; return o; };

  half_t* enc_in16 = (half_t*)(ws + alloc((size_t)NROWS * DIN * 2));
  half_t* enc16    = (half_t*)(ws + alloc((size_t)NROWS * HDIM * 2));
  float*  gx_fw    = (float*) (ws + alloc((size_t)NROWS * 512 * 4));
  float*  gx_bw    = (float*) (ws + alloc((size_t)NROWS * 512 * 4));
  float*  gtab     = (float*) (ws + alloc((size_t)208 * 768 * 4));
  half_t* ghh16    = (half_t*)(ws + alloc((size_t)768 * 256 * 2));
  half_t* proj16   = (half_t*)(ws + alloc((size_t)208 * 256 * 2));
  half_t* tag16    = (half_t*)(ws + alloc((size_t)640 * 256 * 2));   // padded to 640
  half_t* init16   = (half_t*)(ws + alloc((size_t)256 * 256 * 2));
  half_t* wihf16   = (half_t*)(ws + alloc((size_t)512 * 192 * 2));
  half_t* wihb16   = (half_t*)(ws + alloc((size_t)512 * 192 * 2));
  half_t* whhf16   = (half_t*)(ws + alloc((size_t)512 * 128 * 2));
  half_t* whhb16   = (half_t*)(ws + alloc((size_t)512 * 128 * 2));
  half_t* lemb16   = (half_t*)(ws + alloc((size_t)208 * 64 * 2));
  half_t* gwih16   = (half_t*)(ws + alloc((size_t)768 * 64 * 2));
  float*  projbp   = (float*) (ws + alloc((size_t)208 * 4));

  auto cvt = [&](const float* s, half_t* d, int r, int c, int sr, int sc) {
    int cnt = r * c;
    cvt_pad16_k<<<(cnt + 255) / 256, 256, 0, stream>>>(s, d, r, c, sr, sc);
  };
  cvt(gru_whh, ghh16, 768, 256, 768, 256);
  cvt(proj_w,  proj16, 208, 256, 200, 256);
  cvt(tag_w,   tag16,  640, 256, 600, 256);
  cvt(init_w,  init16, 256, 256, 256, 256);
  cvt(fw_wih,  wihf16, 512, 192, 512, 192);
  cvt(bw_wih,  wihb16, 512, 192, 512, 192);
  cvt(fw_whh,  whhf16, 512, 128, 512, 128);
  cvt(bw_whh,  whhb16, 512, 128, 512, 128);
  cvt(lemma_emb, lemb16, 208, 64, 200, 48);
  cvt(gru_wih, gwih16, 768, 64, 768, 48);
  pad_bias_k<<<1, 256, 0, stream>>>(proj_b, projbp, 208, 200, -1e30f);

  // embeddings + char CNN
  embed_conv_k<<<NROWS, 64, 0, stream>>>(word_ids, char_ids, word_emb, char_emb,
                                         conv_w, conv_b, enc_in16);

  // Gx = enc_in @ Wih^T + b   (both directions)  [Npad=512 -> 8 groups of 64]
  wmma_gemm_k<<<(NROWS / 16) * (512 / 64), 32, 0, stream>>>(
      enc_in16, wihf16, fw_b, gx_fw, NROWS, 512, DIN, 512, 512);
  wmma_gemm_k<<<(NROWS / 16) * (512 / 64), 32, 0, stream>>>(
      enc_in16, wihb16, bw_b, gx_bw, NROWS, 512, DIN, 512, 512);

  // Gtab = lemma_emb @ gru_wih^T + bih  (per-token GRU input precompute)
  wmma_gemm_k<<<(208 / 16) * (768 / 64), 32, 0, stream>>>(
      lemb16, gwih16, gru_bih, gtab, 208, 768, 64, 768, 768);

  // BiLSTM recurrence -> enc16
  lstm_rec_k<<<8, 128, 0, stream>>>(whhf16, whhb16, gx_fw, gx_bw, enc16);

  // tag logits straight into d_out  [Npad=640 -> 10 groups of 64]
  float* out = (float*)d_out;
  wmma_gemm_k<<<(NROWS / 16) * (640 / 64), 32, 0, stream>>>(
      enc16, tag16, tag_b, out, NROWS, 640, HDIM, TAGS, TAGS);

  // fused greedy lemma decoder
  float* out_lemma = out + (size_t)NROWS * TAGS;
  decoder_k<<<NROWS / 16, 128, 0, stream>>>(enc16, init16, init_b, ghh16, gru_bhh,
                                            gtab, proj16, projbp, out_lemma);
}
